// EgyptianFractionAttention_81982335746059
// MI455X (gfx1250) — matrix operand order repaired and checked by reading, not compiled, not run on hardware
//
#include <hip/hip_runtime.h>
#include <hip/hip_bf16.h>

typedef __attribute__((ext_vector_type(16))) _Float16 v16h;
typedef __attribute__((ext_vector_type(8)))  _Float16 v8h;
typedef __attribute__((ext_vector_type(8)))  float    v8f;
typedef __attribute__((ext_vector_type(4)))  unsigned int v4u;
typedef __attribute__((ext_vector_type(8)))  int v8i;
typedef __attribute__((ext_vector_type(4)))  int v4i;

#define B_    2
#define S_    2048
#define D_    1536
#define H_    12
#define HD_   128
#define N3_   4608          // 3*D_
#define ROWS_ 4096          // B_*S_
#define SCALE_ 0.08838834764831845f   // 128^-0.5

static __device__ __forceinline__ v16h make16(v8h lo, v8h hi) {
  v16h r;
#pragma unroll
  for (int i = 0; i < 8; ++i) { r[i] = lo[i]; r[8 + i] = hi[i]; }
  return r;
}

// Issue one TDM 2-D tile load Global->LDS (CDNA5 tensor_load_to_lds).
// D# built per CDNA5 ISA. All args block-uniform.
// This toolchain exposes the 6-arg builtin (trailing int32x8 = extra
// descriptor words for >2-D / gather modes; zero here).
static __device__ __forceinline__ void tdm_load_2d(
    unsigned lds_off, const _Float16* gptr,
    unsigned tensor_d0, unsigned tensor_d1,
    unsigned tile_d0, unsigned tile_d1, unsigned d0_stride) {
  unsigned long long ga = (unsigned long long)(size_t)gptr;
  v4u g0;
  g0[0] = 1u;                                         // count=1 (valid descriptor)
  g0[1] = lds_off;                                    // LDS byte address
  g0[2] = (unsigned)(ga & 0xffffffffu);               // global_addr[31:0]
  g0[3] = (unsigned)((ga >> 32) & 0x1ffffffu) | (2u << 30);  // addr[56:32] | type=2
  v8i g1;
  g1[0] = (int)(1u << 16);                            // data_size=1 -> 2 bytes
  g1[1] = (int)((tensor_d0 & 0xffffu) << 16);         // tensor_dim0[15:0]
  g1[2] = (int)(((tensor_d0 >> 16) & 0xffffu) |
                ((tensor_d1 & 0xffffu) << 16));       // dim0[31:16] | dim1[15:0]
  g1[3] = (int)(((tensor_d1 >> 16) & 0xffffu) |
                ((tile_d0 & 0xffffu) << 16));         // dim1[31:16] | tile_dim0
  g1[4] = (int)(tile_d1 & 0xffffu);                   // tile_dim1 (tile_dim2=0)
  g1[5] = (int)d0_stride;                             // tensor_dim0_stride[31:0]
  g1[6] = 0;
  g1[7] = 0;
  v4i g2 = {0, 0, 0, 0};
  v4i g3 = {0, 0, 0, 0};
  v8i g4 = {0, 0, 0, 0, 0, 0, 0, 0};
  __builtin_amdgcn_tensor_load_to_lds(g0, g1, g2, g3, g4, 0);
}

static __device__ __forceinline__ unsigned lds_offset_of(const void* p) {
  // LDS aperture: flat addr[31:0] is the LDS byte offset.
  return (unsigned)(size_t)p;
}

// ---------------------------------------------------------------- conversions
__global__ void k_cvt_x(const float* __restrict__ x, _Float16* __restrict__ xh, int n) {
  int i = blockIdx.x * blockDim.x + threadIdx.x;
  if (i < n) xh[i] = (_Float16)x[i];
}

// Wt[c*K + k] = (f16) W[k*N + c]   (column-major f16 copy of W[K,N])
__global__ void k_transpose_f16(const float* __restrict__ W, _Float16* __restrict__ Wt,
                                int K, int N) {
  long long i = (long long)blockIdx.x * blockDim.x + threadIdx.x;
  long long total = (long long)K * N;
  if (i >= total) return;
  int c = (int)(i / K), k = (int)(i % K);
  Wt[i] = (_Float16)W[(long long)k * N + c];
}

// ---------------------------------------------------------------- QKV GEMM
__global__ void __launch_bounds__(256) k_gemm_qkv(
    const _Float16* __restrict__ A, const _Float16* __restrict__ Bt,
    const float* __restrict__ bias,
    _Float16* __restrict__ Qb, _Float16* __restrict__ Kb, _Float16* __restrict__ Vt) {
  const int lane = threadIdx.x & 31, wid = threadIdx.x >> 5;
  const int half = lane >> 4, l16 = lane & 15;
  const int m0 = blockIdx.y * 128 + wid * 16;
  const int n0 = blockIdx.x * 64;

  v8f acc[4] = {};
  const _Float16* arow = A + (size_t)(m0 + l16) * D_;
  for (int k = 0; k < D_; k += 32) {
    __builtin_prefetch(arow + k + 256, 0, 0);
    v8h a0 = *(const v8h*)(arow + k + half * 16);
    v8h a1 = *(const v8h*)(arow + k + half * 16 + 8);
    v16h af = make16(a0, a1);
#pragma unroll
    for (int s = 0; s < 4; ++s) {
      const _Float16* bp = Bt + (size_t)(n0 + s * 16 + l16) * D_ + k + half * 16;
      v16h bf = *(const v16h*)bp;
      acc[s] = __builtin_amdgcn_wmma_f32_16x16x32_f16(false, af, false, bf,
                                                      (short)0, acc[s], false, false);
    }
  }
#pragma unroll
  for (int s = 0; s < 4; ++s) {
    int c = n0 + s * 16 + l16;
    int t = c / D_, rem = c % D_;
    int h = rem / HD_, d = rem % HD_;
    float bc = bias[c];
#pragma unroll
    for (int r = 0; r < 8; ++r) {
      int row = m0 + r + 8 * half;
      int b = row >> 11, srow = row & (S_ - 1);
      float v = acc[s][r] + bc;
      size_t bh = (size_t)(b * H_ + h);
      if (t == 0)      Qb[(bh * S_ + srow) * HD_ + d] = (_Float16)(v * SCALE_);
      else if (t == 1) Kb[(bh * S_ + srow) * HD_ + d] = (_Float16)v;
      else             Vt[(bh * HD_ + d) * S_ + srow] = (_Float16)v;
    }
  }
}

// ---------------------------------------------------------------- output GEMM
__global__ void __launch_bounds__(256) k_gemm_out(
    const _Float16* __restrict__ A, const _Float16* __restrict__ Bt,
    const float* __restrict__ bias, float* __restrict__ out) {
  const int lane = threadIdx.x & 31, wid = threadIdx.x >> 5;
  const int half = lane >> 4, l16 = lane & 15;
  const int m0 = blockIdx.y * 128 + wid * 16;
  const int n0 = blockIdx.x * 64;

  v8f acc[4] = {};
  const _Float16* arow = A + (size_t)(m0 + l16) * D_;
  for (int k = 0; k < D_; k += 32) {
    __builtin_prefetch(arow + k + 256, 0, 0);
    v8h a0 = *(const v8h*)(arow + k + half * 16);
    v8h a1 = *(const v8h*)(arow + k + half * 16 + 8);
    v16h af = make16(a0, a1);
#pragma unroll
    for (int s = 0; s < 4; ++s) {
      const _Float16* bp = Bt + (size_t)(n0 + s * 16 + l16) * D_ + k + half * 16;
      v16h bf = *(const v16h*)bp;
      acc[s] = __builtin_amdgcn_wmma_f32_16x16x32_f16(false, af, false, bf,
                                                      (short)0, acc[s], false, false);
    }
  }
#pragma unroll
  for (int s = 0; s < 4; ++s) {
    int n = n0 + s * 16 + l16;
    float bc = bias[n];
#pragma unroll
    for (int r = 0; r < 8; ++r) {
      int row = m0 + r + 8 * half;
      out[(size_t)row * D_ + n] = acc[s][r] + bc;
    }
  }
}

// ---------------------------------------------------------------- full-attention
// One block = one (b, head) x 128 consecutive queries (8 waves x 16-query tiles).
// K/V 32-key tiles are double-buffered in LDS via the Tensor Data Mover
// (wave 0 issues tensor_load_to_lds, s_wait_tensorcnt gates the pipeline).
__global__ void __launch_bounds__(256) k_attn_full(
    const _Float16* __restrict__ Qb, const _Float16* __restrict__ Kb,
    const _Float16* __restrict__ Vt, _Float16* __restrict__ attn,
    int heads_per_b) {
  __shared__ __align__(32) _Float16 Ksh[2][32 * HD_];   // [key][d]
  __shared__ __align__(32) _Float16 Vsh[2][HD_ * 32];   // [d][key]
  __shared__ __align__(32) _Float16 Plds[8 * 16 * 32];

  const int lane = threadIdx.x & 31, wid = threadIdx.x >> 5;
  const int half = lane >> 4, l16 = lane & 15;

  int qb = blockIdx.x & 15;              // 16 query-blocks of 128
  int bhid = blockIdx.x >> 4;
  int h = bhid % heads_per_b;
  int b = bhid / heads_per_b;
  const int q0 = (qb * 8 + wid) * 16;
  const size_t bh = (size_t)(b * H_ + h);
  const _Float16* Qp = Qb + bh * S_ * HD_;
  const _Float16* Kp = Kb + bh * S_ * HD_;
  const _Float16* Vp = Vt + bh * HD_ * S_;
  _Float16* Pw = &Plds[wid * 512];

  v16h qf[4];
  {
    const _Float16* qr = Qp + (size_t)(q0 + l16) * HD_;
#pragma unroll
    for (int c = 0; c < 4; ++c) {
      v8h a0 = *(const v8h*)(qr + c * 32 + half * 16);
      v8h a1 = *(const v8h*)(qr + c * 32 + half * 16 + 8);
      qf[c] = make16(a0, a1);
    }
  }
  v8f acc[8] = {};
  float mst[8], lst[8];
#pragma unroll
  for (int r = 0; r < 8; ++r) { mst[r] = -3.0e38f; lst[r] = 0.f; }

  const int NT = S_ / 32;                // 64 key tiles
  if (wid == 0) {
    // preload tile 0 into buffer 0
    tdm_load_2d(lds_offset_of(&Ksh[0][0]), Kp, HD_, S_, HD_, 32, HD_);
    tdm_load_2d(lds_offset_of(&Vsh[0][0]), Vp, S_, HD_, 32, HD_, S_);
  }

  for (int it = 0; it < NT; ++it) {
    const int buf = it & 1;
    if (wid == 0) {
      if (it + 1 < NT) {
        const int j1 = (it + 1) * 32;
        tdm_load_2d(lds_offset_of(&Ksh[buf ^ 1][0]), Kp + (size_t)j1 * HD_,
                    HD_, S_, HD_, 32, HD_);
        tdm_load_2d(lds_offset_of(&Vsh[buf ^ 1][0]), Vp + j1,
                    S_, HD_, 32, HD_, S_);
        __builtin_amdgcn_s_wait_tensorcnt(2);   // tile `it` complete (in-order)
      } else {
        __builtin_amdgcn_s_wait_tensorcnt(0);
      }
    }
    __syncthreads();                            // tiles visible to all waves

    v8f sA = {}, sB = {};
#pragma unroll
    for (int c = 0; c < 4; ++c) {
      v16h k0 = *(const v16h*)(&Ksh[buf][(l16) * HD_ + c * 32 + half * 16]);
      v16h k1 = *(const v16h*)(&Ksh[buf][(16 + l16) * HD_ + c * 32 + half * 16]);
      sA = __builtin_amdgcn_wmma_f32_16x16x32_f16(false, qf[c], false, k0,
                                                  (short)0, sA, false, false);
      sB = __builtin_amdgcn_wmma_f32_16x16x32_f16(false, qf[c], false, k1,
                                                  (short)0, sB, false, false);
    }
#pragma unroll
    for (int r = 0; r < 8; ++r) {
      float rm = fmaxf(sA[r], sB[r]);
#pragma unroll
      for (int off = 1; off < 16; off <<= 1) rm = fmaxf(rm, __shfl_xor(rm, off, 32));
      float mnew = fmaxf(mst[r], rm);
      float alpha = __expf(mst[r] - mnew);
      float pA = __expf(sA[r] - mnew);
      float pB = __expf(sB[r] - mnew);
      float rs = pA + pB;
#pragma unroll
      for (int off = 1; off < 16; off <<= 1) rs += __shfl_xor(rs, off, 32);
      lst[r] = lst[r] * alpha + rs;
      mst[r] = mnew;
#pragma unroll
      for (int dt = 0; dt < 8; ++dt) acc[dt][r] *= alpha;
      int m = r + 8 * half;
      Pw[m * 32 + l16]      = (_Float16)pA;
      Pw[m * 32 + 16 + l16] = (_Float16)pB;
    }
    // Restage P as an A-fragment (same-wave DS ops complete in order).
    v8h p0 = *(const v8h*)(Pw + l16 * 32 + half * 16);
    v8h p1 = *(const v8h*)(Pw + l16 * 32 + half * 16 + 8);
    v16h pf = make16(p0, p1);
#pragma unroll
    for (int dt = 0; dt < 8; ++dt) {
      v16h vf = *(const v16h*)(&Vsh[buf][(dt * 16 + l16) * 32 + half * 16]);
      acc[dt] = __builtin_amdgcn_wmma_f32_16x16x32_f16(false, pf, false, vf,
                                                       (short)0, acc[dt], false, false);
    }
    __syncthreads();                            // done reading buf before TDM reuses it
  }

#pragma unroll
  for (int r = 0; r < 8; ++r) {
    int row = b * S_ + q0 + r + 8 * half;
    float inv = 1.0f / lst[r];
#pragma unroll
    for (int dt = 0; dt < 8; ++dt)
      attn[(size_t)row * D_ + h * HD_ + dt * 16 + l16] = (_Float16)(acc[dt][r] * inv);
  }
}

// ---------------------------------------------------------------- local (windowed) heads
// Per-wave key ranges differ, so keep direct-load flash kernel with masking.
__global__ void __launch_bounds__(256) k_attn_local(
    const _Float16* __restrict__ Qb, const _Float16* __restrict__ Kb,
    const _Float16* __restrict__ Vt, _Float16* __restrict__ attn,
    int head_base, int heads_per_b) {
  __shared__ __align__(32) _Float16 Plds[8 * 16 * 32];
  const int lane = threadIdx.x & 31, wid = threadIdx.x >> 5;
  const int half = lane >> 4, l16 = lane & 15;

  int task = blockIdx.x * 8 + wid;
  int qt = task & 127; task >>= 7;
  int h = head_base + task % heads_per_b;
  int b = task / heads_per_b;
  const int q0 = qt * 16;
  const size_t bh = (size_t)(b * H_ + h);
  const _Float16* Qp = Qb + bh * S_ * HD_;
  const _Float16* Kp = Kb + bh * S_ * HD_;
  const _Float16* Vp = Vt + bh * HD_ * S_;
  _Float16* Pw = &Plds[wid * 512];

  v16h qf[4];
  {
    const _Float16* qr = Qp + (size_t)(q0 + l16) * HD_;
#pragma unroll
    for (int c = 0; c < 4; ++c) {
      v8h a0 = *(const v8h*)(qr + c * 32 + half * 16);
      v8h a1 = *(const v8h*)(qr + c * 32 + half * 16 + 8);
      qf[c] = make16(a0, a1);
    }
  }
  v8f acc[8] = {};
  float mst[8], lst[8];
#pragma unroll
  for (int r = 0; r < 8; ++r) { mst[r] = -3.0e38f; lst[r] = 0.f; }

  int lo = q0 - 32; if (lo < 0) lo = 0;
  int jstart = lo & ~31;
  int jend = (q0 + 48 + 31) & ~31; if (jend > S_) jend = S_;

  for (int j0 = jstart; j0 < jend; j0 += 32) {
    v8f sA = {}, sB = {};
#pragma unroll
    for (int c = 0; c < 4; ++c) {
      v16h k0 = *(const v16h*)(Kp + (size_t)(j0 + l16) * HD_ + c * 32 + half * 16);
      v16h k1 = *(const v16h*)(Kp + (size_t)(j0 + 16 + l16) * HD_ + c * 32 + half * 16);
      sA = __builtin_amdgcn_wmma_f32_16x16x32_f16(false, qf[c], false, k0,
                                                  (short)0, sA, false, false);
      sB = __builtin_amdgcn_wmma_f32_16x16x32_f16(false, qf[c], false, k1,
                                                  (short)0, sB, false, false);
    }
#pragma unroll
    for (int r = 0; r < 8; ++r) {
      int i = q0 + r + 8 * half;
      int dA = i - (j0 + l16);      if (dA < 0) dA = -dA;
      int dB = i - (j0 + 16 + l16); if (dB < 0) dB = -dB;
      if (dA > 32) sA[r] = -3.0e38f;
      if (dB > 32) sB[r] = -3.0e38f;
    }
#pragma unroll
    for (int r = 0; r < 8; ++r) {
      float rm = fmaxf(sA[r], sB[r]);
#pragma unroll
      for (int off = 1; off < 16; off <<= 1) rm = fmaxf(rm, __shfl_xor(rm, off, 32));
      float mnew = fmaxf(mst[r], rm);
      float alpha = __expf(mst[r] - mnew);
      float pA = __expf(sA[r] - mnew);
      float pB = __expf(sB[r] - mnew);
      float rs = pA + pB;
#pragma unroll
      for (int off = 1; off < 16; off <<= 1) rs += __shfl_xor(rs, off, 32);
      lst[r] = lst[r] * alpha + rs;
      mst[r] = mnew;
#pragma unroll
      for (int dt = 0; dt < 8; ++dt) acc[dt][r] *= alpha;
      int m = r + 8 * half;
      Pw[m * 32 + l16]      = (_Float16)pA;
      Pw[m * 32 + 16 + l16] = (_Float16)pB;
    }
    v8h p0 = *(const v8h*)(Pw + l16 * 32 + half * 16);
    v8h p1 = *(const v8h*)(Pw + l16 * 32 + half * 16 + 8);
    v16h pf = make16(p0, p1);
#pragma unroll
    for (int dt = 0; dt < 8; ++dt) {
      v16h vf = *(const v16h*)(Vp + (size_t)(dt * 16 + l16) * S_ + j0 + half * 16);
      acc[dt] = __builtin_amdgcn_wmma_f32_16x16x32_f16(false, pf, false, vf,
                                                       (short)0, acc[dt], false, false);
    }
  }

#pragma unroll
  for (int r = 0; r < 8; ++r) {
    int row = b * S_ + q0 + r + 8 * half;
    float inv = 1.0f / lst[r];
#pragma unroll
    for (int dt = 0; dt < 8; ++dt)
      attn[(size_t)row * D_ + h * HD_ + dt * 16 + l16] = (_Float16)(acc[dt][r] * inv);
  }
}

// ---------------------------------------------------------------- global heads (2 keys)
__global__ void __launch_bounds__(256) k_attn_global(
    const _Float16* __restrict__ Qb, const _Float16* __restrict__ Kb,
    const _Float16* __restrict__ Vt, _Float16* __restrict__ attn) {
  const int lane = threadIdx.x & 31, wid = threadIdx.x >> 5;
  int task = blockIdx.x * 8 + wid;               // (b, hg, q)
  int q = task & (S_ - 1); task >>= 11;
  int hg = task & 1; int b = task >> 1;
  int h = H_ - 2 + hg;
  size_t bh = (size_t)(b * H_ + h);
  const _Float16* qp = Qb + (bh * S_ + q) * HD_ + lane * 4;
  const _Float16* k0 = Kb + (bh * S_ + 0) * HD_ + lane * 4;
  const _Float16* k1 = Kb + (bh * S_ + (S_ - 1)) * HD_ + lane * 4;
  float d0 = 0.f, d1 = 0.f;
#pragma unroll
  for (int i = 0; i < 4; ++i) {
    float qv = (float)qp[i];
    d0 += qv * (float)k0[i];
    d1 += qv * (float)k1[i];
  }
#pragma unroll
  for (int off = 1; off < 32; off <<= 1) {
    d0 += __shfl_xor(d0, off, 32);
    d1 += __shfl_xor(d1, off, 32);
  }
  float mx = fmaxf(d0, d1);
  float e0 = __expf(d0 - mx), e1 = __expf(d1 - mx);
  float inv = 1.0f / (e0 + e1);
  size_t row = (size_t)b * S_ + q;
#pragma unroll
  for (int i = 0; i < 4; ++i) {
    int d = lane * 4 + i;
    float v0 = (float)Vt[(bh * HD_ + d) * S_ + 0];
    float v1 = (float)Vt[(bh * HD_ + d) * S_ + (S_ - 1)];
    attn[row * D_ + h * HD_ + d] = (_Float16)((e0 * v0 + e1 * v1) * inv);
  }
}

// ---------------------------------------------------------------- launch
extern "C" void kernel_launch(void* const* d_in, const int* in_sizes, int n_in,
                              void* d_out, int out_size, void* d_ws, size_t ws_size,
                              hipStream_t stream) {
  const float* x    = (const float*)d_in[0];
  const float* Wqkv = (const float*)d_in[1];
  const float* bqkv = (const float*)d_in[2];
  const float* Wout = (const float*)d_in[3];
  const float* bout = (const float*)d_in[4];
  float* out = (float*)d_out;

  char* ws = (char*)d_ws;
  size_t off = 0;
  auto alloc = [&](size_t bytes) -> char* {
    char* p = ws + off;
    off += (bytes + 255) & ~(size_t)255;
    return p;
  };
  _Float16* xh    = (_Float16*)alloc((size_t)ROWS_ * D_ * 2);
  _Float16* WqkvT = (_Float16*)alloc((size_t)N3_ * D_ * 2);
  _Float16* WoutT = (_Float16*)alloc((size_t)D_ * D_ * 2);
  _Float16* Qb    = (_Float16*)alloc((size_t)B_ * H_ * S_ * HD_ * 2);
  _Float16* Kb    = (_Float16*)alloc((size_t)B_ * H_ * S_ * HD_ * 2);
  _Float16* Vt    = (_Float16*)alloc((size_t)B_ * H_ * S_ * HD_ * 2);
  _Float16* attn  = (_Float16*)alloc((size_t)ROWS_ * D_ * 2);

  k_cvt_x<<<(ROWS_ * D_ + 255) / 256, 256, 0, stream>>>(x, xh, ROWS_ * D_);
  k_transpose_f16<<<(int)(((long long)N3_ * D_ + 255) / 256), 256, 0, stream>>>(Wqkv, WqkvT, D_, N3_);
  k_transpose_f16<<<(int)(((long long)D_ * D_ + 255) / 256), 256, 0, stream>>>(Wout, WoutT, D_, D_);

  k_gemm_qkv<<<dim3(N3_ / 64, ROWS_ / 128), 256, 0, stream>>>(xh, WqkvT, bqkv, Qb, Kb, Vt);

  k_attn_full<<<B_ * 6 * 16, 256, 0, stream>>>(Qb, Kb, Vt, attn, 6);         // full heads 0..5
  k_attn_local<<<(B_ * 4 * (S_ / 16)) / 8, 256, 0, stream>>>(Qb, Kb, Vt, attn, 6, 4);
  k_attn_global<<<(B_ * 2 * S_) / 8, 256, 0, stream>>>(Qb, Kb, Vt, attn);

  k_gemm_out<<<dim3(D_ / 64, ROWS_ / 128), 256, 0, stream>>>(attn, WoutT, bout, out);
}
